// SelfAttention_85048942395844
// MI455X (gfx1250) — matrix operand order repaired and checked by reading, not compiled, not run on hardware
//
#include <hip/hip_runtime.h>
#include <hip/hip_bf16.h>

// ---------------------------------------------------------------------------
// Causal multi-head self-attention forward, bf16 WMMA pipeline for gfx1250.
//   qkv = x @ Wqkv        (bf16 WMMA GEMM, async-LDS double buffered)
//   attn = flash-attention (bf16 WMMA QK^T and P*V, f32 online softmax)
//   out  = attn @ Wproj    (same GEMM, f32 out)
// ---------------------------------------------------------------------------

#define N_HEAD 16
#define B_ 4
#define T_ 2048
#define C_ 1024
#define HD 64

typedef __attribute__((ext_vector_type(16))) __bf16 bv16;
typedef __attribute__((ext_vector_type(8)))  float  v8f;

union bvu { bv16 v; uint4 q[2]; };
union v8u { v8f v; float f[8]; };
union u4s { uint4 q; unsigned short s[8]; };

__device__ __forceinline__ unsigned int bf16_1(float f) {
    unsigned int u = __float_as_uint(f);
    return (u + 0x7FFFu + ((u >> 16) & 1u)) >> 16;   // RNE truncation to bf16
}
__device__ __forceinline__ unsigned int pk2(float lo, float hi) {
    return bf16_1(lo) | (bf16_1(hi) << 16);
}

// Async global->LDS 16-byte copy (CDNA5 ASYNCcnt path, bypasses VGPRs).
// vdst = LDS byte address, vaddr = 64-bit global address.
__device__ __forceinline__ void async_copy_b128(const void* gaddr,
                                                void* lds_generic) {
    unsigned int lds_addr = (unsigned int)(uintptr_t)lds_generic;
    asm volatile("global_load_async_to_lds_b128 %0, %1, off"
                 :
                 : "v"(lds_addr), "v"(gaddr)
                 : "memory");
}
__device__ __forceinline__ void async_wait_all() {
    asm volatile("s_wait_asynccnt 0x0" ::: "memory");
}

// ---------------------------------------------------------------------------
// f32 -> bf16 conversion, 8 elements per thread
// ---------------------------------------------------------------------------
__global__ void f32_to_bf16_kernel(const float* __restrict__ in,
                                   unsigned short* __restrict__ out, int n8) {
    int i = blockIdx.x * blockDim.x + threadIdx.x;
    if (i >= n8) return;
    float4 a = ((const float4*)in)[2 * i];
    float4 b = ((const float4*)in)[2 * i + 1];
    uint4 o;
    o.x = pk2(a.x, a.y);
    o.y = pk2(a.z, a.w);
    o.z = pk2(b.x, b.y);
    o.w = pk2(b.z, b.w);
    ((uint4*)out)[i] = o;
}

// ---------------------------------------------------------------------------
// bf16 GEMM: C[M,N] = A[M,K] * B[K,N], f32 accumulate.
// Block = 256 threads = 8 waves (2 in M x 4 in N). Block tile 128x256,
// wave tile 64x64 = 16 WMMA accumulators. K stepped by 32 through LDS.
// Double-buffered: global->LDS staging uses async-LDS engine and overlaps
// with WMMA compute on the other buffer; one barrier per K step.
// ---------------------------------------------------------------------------
template <bool OUT_BF16>
__global__ __launch_bounds__(256)
void gemm_bf16(const unsigned short* __restrict__ A,
               const unsigned short* __restrict__ Bm,
               void* __restrict__ Cout, int M, int N, int K) {
    __shared__ __attribute__((aligned(16))) unsigned short As[2][128 * 32];
    __shared__ __attribute__((aligned(16))) unsigned short Bs[2][32 * 256];

    const int tid  = threadIdx.x;
    const int lane = tid & 31;
    const int w    = tid >> 5;
    const int wm   = w & 1;          // wave position in M (0..1)
    const int wn   = w >> 1;         // wave position in N (0..3)
    const int m0   = blockIdx.y * 128;
    const int n0   = blockIdx.x * 256;
    const int half = lane >> 4;
    const int l15  = lane & 15;

    // Stage one 32-wide K slice into buffer `buf` via async-LDS copies.
    auto stage = [&](int buf, int k0) {
        // A tile: 128 rows x 32 cols bf16 (512 16B chunks, 2 per thread)
#pragma unroll
        for (int c = tid; c < 512; c += 256) {
            int row = c >> 2, cc = c & 3;
            async_copy_b128(&A[(size_t)(m0 + row) * K + k0 + cc * 8],
                            &As[buf][row * 32 + cc * 8]);
        }
        // B tile: 32 rows x 256 cols bf16 (1024 16B chunks, 4 per thread)
#pragma unroll
        for (int c = tid; c < 1024; c += 256) {
            int row = c >> 5, cc = c & 31;
            async_copy_b128(&Bm[(size_t)(k0 + row) * N + n0 + cc * 8],
                            &Bs[buf][row * 256 + cc * 8]);
        }
    };

    v8u acc[4][4];
#pragma unroll
    for (int ms = 0; ms < 4; ++ms)
#pragma unroll
        for (int ns = 0; ns < 4; ++ns)
            acc[ms][ns].v = 0.f;

    stage(0, 0);

    for (int k0 = 0; k0 < K; k0 += 32) {
        const int cur = (k0 >> 5) & 1;
        async_wait_all();    // own wave's loads into `cur` have landed
        __syncthreads();     // all waves' loads landed; prev-buffer reads done
        if (k0 + 32 < K) stage(cur ^ 1, k0 + 32);  // overlap with compute

        // B fragments: lane holds N column (lane&15), 16 K values in table order
        bv16 bfrag[4];
#pragma unroll
        for (int ns = 0; ns < 4; ++ns) {
            bvu u;
            unsigned short* us = (unsigned short*)u.q;
#pragma unroll
            for (int s = 0; s < 16; ++s)
                us[s] = Bs[cur][((s >> 3) * 16 + half * 8 + (s & 7)) * 256 + wn * 64 + ns * 16 + l15];
            bfrag[ns] = u.v;
        }
#pragma unroll
        for (int ms = 0; ms < 4; ++ms) {
            // A fragment: lane holds row (lane&15), two 8-elem K chunks
            bvu ua;
            int row = wm * 64 + ms * 16 + l15;
            ua.q[0] = *(const uint4*)&As[cur][row * 32 + half * 8];
            ua.q[1] = *(const uint4*)&As[cur][row * 32 + 16 + half * 8];
#pragma unroll
            for (int ns = 0; ns < 4; ++ns)
                acc[ms][ns].v = __builtin_amdgcn_wmma_f32_16x16x32_bf16(
                    false, ua.v, false, bfrag[ns], (short)0, acc[ms][ns].v,
                    false, false);
        }
    }

    // Epilogue: lane holds column (lane&15), rows r + 8*(lane>>4)
#pragma unroll
    for (int ms = 0; ms < 4; ++ms)
#pragma unroll
        for (int ns = 0; ns < 4; ++ns) {
            int ng = n0 + wn * 64 + ns * 16 + l15;
#pragma unroll
            for (int r = 0; r < 8; ++r) {
                int mg = m0 + wm * 64 + ms * 16 + half * 8 + r;
                float val = acc[ms][ns].f[r];
                if (OUT_BF16)
                    ((unsigned short*)Cout)[(size_t)mg * N + ng] =
                        (unsigned short)bf16_1(val);
                else
                    ((float*)Cout)[(size_t)mg * N + ng] = val;
            }
        }
}

// ---------------------------------------------------------------------------
// Flash attention: block = 128 threads (4 waves), one 64-row Q tile per block
// per (b,h). Wave handles 16 query rows; loops over 64-key tiles kt<=qt.
// S and O accumulate in f32 WMMA registers; online softmax via exp2f.
// ---------------------------------------------------------------------------
__global__ __launch_bounds__(128)
void attn_kernel(const unsigned short* __restrict__ qkv,
                 unsigned short* __restrict__ out) {
    __shared__ __attribute__((aligned(16))) unsigned short Qs[64 * 64];
    __shared__ __attribute__((aligned(16))) unsigned short Kts[64 * 64]; // K^T: [d][key]
    __shared__ __attribute__((aligned(16))) unsigned short Vs[64 * 64];  // [key][d]
    __shared__ __attribute__((aligned(16))) unsigned short Ps[4][16 * 64];

    const int tid  = threadIdx.x;
    const int lane = tid & 31;
    const int w    = tid >> 5;
    const int half = lane >> 4;
    const int l15  = lane & 15;
    const int qt   = blockIdx.x;
    const int bh   = blockIdx.y;
    const int b    = bh >> 4;
    const int h    = bh & 15;

    const size_t rs = 3 * C_;  // qkv row stride in elements
    const unsigned short* qbase = qkv + (size_t)b * T_ * rs + h * HD;
    const unsigned short* kbase = qbase + C_;
    const unsigned short* vbase = qbase + 2 * C_;

    // Stage Q tile (64x64)
#pragma unroll
    for (int c = tid; c < 512; c += 128) {
        int r = c >> 3, cc = c & 7;
        *(uint4*)&Qs[r * 64 + cc * 8] =
            *(const uint4*)&qbase[(size_t)(qt * 64 + r) * rs + cc * 8];
    }
    __syncthreads();

    // Q A-fragments (two 32-wide d chunks), loaded once
    bv16 qf[2];
    {
        int row = w * 16 + l15;
#pragma unroll
        for (int dd = 0; dd < 2; ++dd) {
            bvu u;
            u.q[0] = *(const uint4*)&Qs[row * 64 + dd * 32 + half * 8];
            u.q[1] = *(const uint4*)&Qs[row * 64 + dd * 32 + 16 + half * 8];
            qf[dd] = u.v;
        }
    }

    const float NEGBIG = -1e30f;
    v8u oacc[4];
    float m_i[8], l_i[8];
#pragma unroll
    for (int ds = 0; ds < 4; ++ds) oacc[ds].v = 0.f;
#pragma unroll
    for (int r = 0; r < 8; ++r) { m_i[r] = NEGBIG; l_i[r] = 0.f; }

    const float sscale = 0.125f * 1.44269504088896340736f;  // log2(e)/sqrt(hd)

    for (int kt = 0; kt <= qt; ++kt) {
        __syncthreads();
        // Stage K^T (transpose on LDS write) and V
#pragma unroll
        for (int c = tid; c < 512; c += 128) {
            int r = c >> 3, cc = c & 7;  // key row r, d chunk cc
            u4s kv;
            kv.q = *(const uint4*)&kbase[(size_t)(kt * 64 + r) * rs + cc * 8];
#pragma unroll
            for (int j = 0; j < 8; ++j) Kts[(cc * 8 + j) * 64 + r] = kv.s[j];
            *(uint4*)&Vs[r * 64 + cc * 8] =
                *(const uint4*)&vbase[(size_t)(kt * 64 + r) * rs + cc * 8];
        }
        __syncthreads();

        // S = Q K^T  (16x64 per wave, 4 accumulators)
        v8u sacc[4];
#pragma unroll
        for (int ns = 0; ns < 4; ++ns) {
            sacc[ns].v = 0.f;
#pragma unroll
            for (int dd = 0; dd < 2; ++dd) {
                bvu u;
                unsigned short* us = (unsigned short*)u.q;
#pragma unroll
                for (int s = 0; s < 16; ++s)
                    us[s] = Kts[(dd * 32 + (s >> 3) * 16 + half * 8 + (s & 7)) * 64 + ns * 16 + l15];
                sacc[ns].v = __builtin_amdgcn_wmma_f32_16x16x32_bf16(
                    false, qf[dd], false, u.v, (short)0, sacc[ns].v, false,
                    false);
            }
        }

        // Online softmax (per lane: 8 rows of its 16-lane half)
        const bool diag = (kt == qt);
#pragma unroll
        for (int r = 0; r < 8; ++r) {
            int qrow = qt * 64 + w * 16 + half * 8 + r;
            float mx = NEGBIG;
#pragma unroll
            for (int ns = 0; ns < 4; ++ns) {
                float sv = sacc[ns].f[r] * sscale;
                if (diag) {
                    int kcol = kt * 64 + ns * 16 + l15;
                    if (kcol > qrow) sv = NEGBIG;
                }
                sacc[ns].f[r] = sv;
                mx = fmaxf(mx, sv);
            }
#pragma unroll
            for (int off = 1; off < 16; off <<= 1)
                mx = fmaxf(mx, __shfl_xor(mx, off, 32));
            float mnew = fmaxf(m_i[r], mx);
            float alpha = exp2f(m_i[r] - mnew);
            l_i[r] *= alpha;
#pragma unroll
            for (int ds = 0; ds < 4; ++ds) oacc[ds].f[r] *= alpha;
            float rsum = 0.f;
#pragma unroll
            for (int ns = 0; ns < 4; ++ns) {
                float p = exp2f(sacc[ns].f[r] - mnew);
                sacc[ns].f[r] = p;
                rsum += p;
            }
#pragma unroll
            for (int off = 1; off < 16; off <<= 1)
                rsum += __shfl_xor(rsum, off, 32);
            l_i[r] += rsum;
            m_i[r] = mnew;
        }

        // P (C-layout) -> bf16 -> per-wave LDS slab -> A-layout fragments
#pragma unroll
        for (int ns = 0; ns < 4; ++ns)
#pragma unroll
            for (int r = 0; r < 8; ++r)
                Ps[w][(half * 8 + r) * 64 + ns * 16 + l15] =
                    (unsigned short)bf16_1(sacc[ns].f[r]);
        asm volatile("" ::: "memory");  // keep LDS write->read order

        // O += P * V
#pragma unroll
        for (int kk = 0; kk < 2; ++kk) {
            bvu pa;
            pa.q[0] = *(const uint4*)&Ps[w][l15 * 64 + kk * 32 + half * 8];
            pa.q[1] = *(const uint4*)&Ps[w][l15 * 64 + kk * 32 + 16 + half * 8];
#pragma unroll
            for (int ds = 0; ds < 4; ++ds) {
                bvu vb;
                unsigned short* vs = (unsigned short*)vb.q;
#pragma unroll
                for (int s = 0; s < 16; ++s)
                    vs[s] = Vs[(kk * 32 + (s >> 3) * 16 + half * 8 + (s & 7)) * 64 + ds * 16 + l15];
                oacc[ds].v = __builtin_amdgcn_wmma_f32_16x16x32_bf16(
                    false, pa.v, false, vb.v, (short)0, oacc[ds].v, false,
                    false);
            }
        }
    }

    // Normalize and write attn output as bf16 [B*T, C]
#pragma unroll
    for (int ds = 0; ds < 4; ++ds)
#pragma unroll
        for (int r = 0; r < 8; ++r) {
            int qrow = qt * 64 + w * 16 + half * 8 + r;
            float val = oacc[ds].f[r] / l_i[r];
            out[(size_t)(b * T_ + qrow) * C_ + h * HD + ds * 16 + l15] =
                (unsigned short)bf16_1(val);
        }
}

// ---------------------------------------------------------------------------
extern "C" void kernel_launch(void* const* d_in, const int* in_sizes, int n_in,
                              void* d_out, int out_size, void* d_ws,
                              size_t ws_size, hipStream_t stream) {
    (void)in_sizes; (void)n_in; (void)out_size; (void)ws_size;
    const float* x     = (const float*)d_in[0];
    const float* Wqkv  = (const float*)d_in[1];
    const float* Wproj = (const float*)d_in[2];
    float* out = (float*)d_out;

    const int M  = B_ * T_;   // 8192
    const int K  = C_;        // 1024
    const int N3 = 3 * C_;    // 3072

    unsigned short* xb     = (unsigned short*)d_ws;
    unsigned short* wqkvb  = xb     + (size_t)M * K;
    unsigned short* wprojb = wqkvb  + (size_t)K * N3;
    unsigned short* qkvb   = wprojb + (size_t)K * C_;
    unsigned short* attnb  = qkvb   + (size_t)M * N3;

    // f32 -> bf16 converts
    {
        size_t n8;
        n8 = (size_t)M * K / 8;
        f32_to_bf16_kernel<<<dim3((n8 + 255) / 256), dim3(256), 0, stream>>>(
            x, xb, (int)n8);
        n8 = (size_t)K * N3 / 8;
        f32_to_bf16_kernel<<<dim3((n8 + 255) / 256), dim3(256), 0, stream>>>(
            Wqkv, wqkvb, (int)n8);
        n8 = (size_t)K * C_ / 8;
        f32_to_bf16_kernel<<<dim3((n8 + 255) / 256), dim3(256), 0, stream>>>(
            Wproj, wprojb, (int)n8);
    }

    // qkv = x @ Wqkv (bf16 out)
    gemm_bf16<true><<<dim3(N3 / 256, M / 128), dim3(256), 0, stream>>>(
        xb, wqkvb, (void*)qkvb, M, N3, K);

    // flash attention
    attn_kernel<<<dim3(T_ / 64, B_ * N_HEAD), dim3(128), 0, stream>>>(
        qkvb, attnb);

    // out = attn @ Wproj (f32 out)
    gemm_bf16<false><<<dim3(C_ / 256, M / 128), dim3(256), 0, stream>>>(
        attnb, wprojb, (void*)out, M, C_, K);
}
